// DirectVoxGO_1305670057984
// MI455X (gfx1250) — matrix operand order repaired and checked by reading, not compiled, not run on hardware
//
#include <hip/hip_runtime.h>
#include <math.h>
#include <stdint.h>

typedef __attribute__((ext_vector_type(16))) _Float16 v16h;
typedef __attribute__((ext_vector_type(8)))  float    v8f;

#define WORLD       160
#define NSAMP       256
#define NRAYS       4096
#define XYZ_MIN_F   (-1.0f)
#define STEP_DIST   0.00625f            /* STEPSIZE * VOXEL_SIZE = 0.5 * (2/160) */
#define NEAR_F      0.1f
#define INTERVAL_F  0.5f
#define ACT_SHIFT_F (-13.815510558f)    /* log(1/(1-1e-6) - 1) */
#define THRES_F     1e-4f
#define BG_F        1.0f

#define SXX (WORLD*WORLD)
#define SYY (WORLD)
#define CHVOL (WORLD*WORLD*WORLD)

/* workspace layout (in _Float16 units) */
#define W0T_OFF 0                         /* [128][96]  : W0^T, K padded 72->96 */
#define W1T_OFF (128*96)                  /* [128][128] : W1^T                  */
#define W2T_OFF (128*96 + 128*128)        /* [16][128]  : W2^T, N padded 3->16  */
#define PACK_TOTAL (128*96 + 128*128 + 16*128)

// ---------------------------------------------------------------------------
// Weight pack: f32 row-major [K][N] -> f16 [N][K] (so B fragments are two
// contiguous 16B loads per lane), with zero padding.
// ---------------------------------------------------------------------------
__global__ void pack_weights_kernel(const float* __restrict__ w0,
                                    const float* __restrict__ w1,
                                    const float* __restrict__ w2,
                                    _Float16* __restrict__ ws)
{
    int i = blockIdx.x * blockDim.x + threadIdx.x;
    if (i < 128*96) {
        int n = i / 96, k = i % 96;
        ws[W0T_OFF + i] = (k < 72) ? (_Float16)w0[k*128 + n] : (_Float16)0.0f;
    } else if (i < 128*96 + 128*128) {
        int j = i - 128*96;
        int n = j / 128, k = j % 128;
        ws[W1T_OFF + j] = (_Float16)w1[k*128 + n];
    } else if (i < PACK_TOTAL) {
        int j = i - (128*96 + 128*128);
        int n = j / 128, k = j % 128;
        ws[W2T_OFF + j] = (n < 3) ? (_Float16)w2[k*3 + n] : (_Float16)0.0f;
    }
}

// ---------------------------------------------------------------------------
// WMMA helpers (CDNA5 wave32, V_WMMA_F32_16X16X32_F16)
// ---------------------------------------------------------------------------
__device__ __forceinline__ v8f wmma_f16(v16h a, v16h b, v8f c) {
    return __builtin_amdgcn_wmma_f32_16x16x32_f16(false, a, false, b,
                                                  (short)0, c, false, false);
}

/* A fragment (16x32 f16) from LDS row-major [row][128] tile.
   Lane L: M = L&15; halves 0..7 = K kbase+off+0..7, halves 8..15 = +16..23,
   off = (L>=16)*8.  (ISA 7.12.2, 16-bit A 16x32 layout.) */
__device__ __forceinline__ v16h load_afrag_lds(const _Float16* S, int m0,
                                               int kbase, int lane) {
    const _Float16* p = S + (m0 + (lane & 15)) * 128 + kbase + ((lane >> 4) << 3);
    union { uint4 q[2]; v16h v; } u;
    u.q[0] = *(const uint4*)(p);
    u.q[1] = *(const uint4*)(p + 16);
    return u.v;
}

/* B fragment (32x16 f16) from LDS W^T laid out [N][K], K-stride ks halves.
   Lane L: N = L&15; halves 0..7 = K kbase+off+0..7, halves 8..15 = +16..23, off = (L>=16)*8. */
__device__ __forceinline__ v16h load_bfrag(const _Float16* Wt, int nbase,
                                           int kbase, int ks, int lane) {
    const _Float16* p = Wt + (nbase + (lane & 15)) * ks + kbase + ((lane >> 4) << 3);
    union { uint4 q[2]; v16h v; } u;
    u.q[0] = *(const uint4*)(p);
    u.q[1] = *(const uint4*)(p + 16);
    return u.v;
}

// ---------------------------------------------------------------------------
// Async global -> LDS staging (GLOBAL_LOAD_ASYNC_TO_LDS_B128, ASYNCcnt).
// GVS mode: mem_addr = SADDR(64) + VADDR(32) ; VDST holds the LDS byte addr.
// ---------------------------------------------------------------------------
__device__ __forceinline__ void async_copy_to_lds(const _Float16* __restrict__ src,
                                                  _Float16* lds_dst, int bytes,
                                                  int tid) {
    unsigned lds_base = (unsigned)(uintptr_t)(void*)lds_dst;   // low 32 bits = LDS offset
    unsigned long long gbase = (unsigned long long)(uintptr_t)src;
    for (int off = tid * 16; off < bytes; off += 256 * 16) {
        unsigned lds_addr = lds_base + (unsigned)off;
        asm volatile("global_load_async_to_lds_b128 %0, %1, %2"
                     :: "v"(lds_addr), "v"(off), "s"(gbase)
                     : "memory");
    }
}

// ---------------------------------------------------------------------------
// Trilinear sampling (align_corners=True, matching reference clipping)
// ---------------------------------------------------------------------------
__device__ __forceinline__ int tri_setup(float px, float py, float pz, float* w) {
    float gx = (px - XYZ_MIN_F) * 0.5f * (float)(WORLD - 1);
    float gy = (py - XYZ_MIN_F) * 0.5f * (float)(WORLD - 1);
    float gz = (pz - XYZ_MIN_F) * 0.5f * (float)(WORLD - 1);
    gx = fminf(fmaxf(gx, 0.0f), (float)(WORLD - 1));
    gy = fminf(fmaxf(gy, 0.0f), (float)(WORLD - 1));
    gz = fminf(fmaxf(gz, 0.0f), (float)(WORLD - 1));
    int ix = (int)gx; if (ix > WORLD - 2) ix = WORLD - 2;
    int iy = (int)gy; if (iy > WORLD - 2) iy = WORLD - 2;
    int iz = (int)gz; if (iz > WORLD - 2) iz = WORLD - 2;
    float fx = gx - (float)ix, fy = gy - (float)iy, fz = gz - (float)iz;
    float mx = 1.0f - fx, my = 1.0f - fy, mz = 1.0f - fz;
    w[0] = mx*my*mz; w[1] = mx*my*fz; w[2] = mx*fy*mz; w[3] = mx*fy*fz;
    w[4] = fx*my*mz; w[5] = fx*my*fz; w[6] = fx*fy*mz; w[7] = fx*fy*fz;
    return (ix * WORLD + iy) * WORLD + iz;
}

__device__ __forceinline__ float tri_fetch(const float* __restrict__ g, int base,
                                           const float* w) {
    return g[base]         * w[0] + g[base + 1]         * w[1]
         + g[base + SYY]   * w[2] + g[base + SYY + 1]   * w[3]
         + g[base + SXX]   * w[4] + g[base + SXX + 1]   * w[5]
         + g[base + SXX + SYY] * w[6] + g[base + SXX + SYY + 1] * w[7];
}

// ---------------------------------------------------------------------------
// Fused per-ray renderer: one workgroup (8 wave32) per ray.
// ---------------------------------------------------------------------------
__global__ __launch_bounds__(256)
void render_kernel(const float* __restrict__ rays_o, const float* __restrict__ rays_d,
                   const float* __restrict__ density, const float* __restrict__ k0,
                   const _Float16* __restrict__ W0t, const float* __restrict__ b0,
                   const _Float16* __restrict__ W1t, const float* __restrict__ b1,
                   const _Float16* __restrict__ W2t, const float* __restrict__ b2,
                   float* __restrict__ out)
{
    __shared__ _Float16 sA[64 * 128];   // feature tile / layer-1 output (16 KB)
    __shared__ _Float16 sB[64 * 128];   // layer-0 output               (16 KB)
    __shared__ _Float16 sW0[128 * 96];  // W0^T                          (24 KB)
    __shared__ _Float16 sW1[128 * 128]; // W1^T                          (32 KB)
    __shared__ _Float16 sW2[16 * 128];  // W2^T                          ( 4 KB)
    __shared__ float sCum[NSAMP];
    __shared__ float sWt[NSAMP];
    __shared__ float sBias0[128];
    __shared__ float sBias1[128];
    __shared__ float sAcc[4];
    __shared__ float sLast;

    const int ray  = blockIdx.x;
    const int tid  = threadIdx.x;
    const int lane = tid & 31;
    const int wave = tid >> 5;

    // Kick off async weight staging first so it overlaps the density gathers.
    async_copy_to_lds(W0t, sW0, 128*96*2,  tid);
    async_copy_to_lds(W1t, sW1, 128*128*2, tid);
    async_copy_to_lds(W2t, sW2, 16*128*2,  tid);

    const float ox = rays_o[ray*3+0], oy = rays_o[ray*3+1], oz = rays_o[ray*3+2];
    const float dx = rays_d[ray*3+0], dy = rays_d[ray*3+1], dz = rays_d[ray*3+2];
    const float rinv = rsqrtf(dx*dx + dy*dy + dz*dz);
    const float vx = dx*rinv, vy = dy*rinv, vz = dz*rinv;

    if (tid < 128) { sBias0[tid] = b0[tid]; sBias1[tid] = b1[tid]; }
    if (tid < 4)   sAcc[tid] = 0.0f;

    // ---- Phase A: per-sample alpha ----------------------------------------
    float alpha;
    {
        float t  = NEAR_F + STEP_DIST * (float)tid;
        float px = ox + dx * t, py = oy + dy * t, pz = oz + dz * t;
        bool outb = (px < -1.0f) | (px > 1.0f) | (py < -1.0f) | (py > 1.0f)
                  | (pz < -1.0f) | (pz > 1.0f);
        float w8[8];
        int base = tri_setup(px, py, pz, w8);
        float dens = tri_fetch(density, base, w8);
        float x  = dens + ACT_SHIFT_F;
        float sp = (x > 20.0f) ? x : log1pf(expf(x));
        alpha = outb ? 0.0f : (1.0f - expf(-sp * INTERVAL_F));
    }

    // ---- Parallel inclusive product scan of (1 - alpha) -------------------
    sCum[tid] = fmaxf(1.0f - alpha, 1e-10f);
    __syncthreads();
    #pragma unroll
    for (int off = 1; off < NSAMP; off <<= 1) {
        float x = sCum[tid];
        float y = (tid >= off) ? sCum[tid - off] : 1.0f;
        __syncthreads();
        sCum[tid] = x * y;
        __syncthreads();
    }
    sWt[tid] = alpha * ((tid == 0) ? 1.0f : sCum[tid - 1]);
    if (tid == NSAMP - 1) sLast = sCum[tid];

    // Make the async-staged weights visible to the whole workgroup.
    asm volatile("s_wait_asynccnt 0" ::: "memory");
    __syncthreads();

    // ---- 4 chunks of 64 samples -------------------------------------------
    #pragma unroll 1
    for (int ch = 0; ch < 4; ++ch) {
        // Feature build: 4 threads per row (k0 ch0-5 | k0 ch6-11 | xyz-PE | view-PE+pad)
        {
            int row  = tid >> 2;
            int part = tid & 3;
            int s = ch * 64 + row;
            float t  = NEAR_F + STEP_DIST * (float)s;
            float px = ox + dx * t, py = oy + dy * t, pz = oz + dz * t;
            _Float16* F = &sA[row * 128];
            if (part <= 1) {
                float w8[8];
                int base = tri_setup(px, py, pz, w8);
                int cbeg = part * 6;
                #pragma unroll
                for (int c = 0; c < 6; ++c)
                    F[cbeg + c] = (_Float16)tri_fetch(k0 + (size_t)(cbeg + c) * CHVOL, base, w8);
            } else if (part == 2) {
                float c3[3] = { (px+1.0f)*0.5f, (py+1.0f)*0.5f, (pz+1.0f)*0.5f };
                F[12] = (_Float16)c3[0]; F[13] = (_Float16)c3[1]; F[14] = (_Float16)c3[2];
                #pragma unroll
                for (int a = 0; a < 3; ++a) {
                    float f = c3[a];
                    #pragma unroll
                    for (int q = 0; q < 5; ++q) {
                        float v = f * (float)(1 << q);
                        F[15 + a*5 + q] = (_Float16)__sinf(v);
                        F[30 + a*5 + q] = (_Float16)__cosf(v);
                    }
                }
            } else {
                float c3[3] = { vx, vy, vz };
                F[45] = (_Float16)vx; F[46] = (_Float16)vy; F[47] = (_Float16)vz;
                #pragma unroll
                for (int a = 0; a < 3; ++a) {
                    float f = c3[a];
                    #pragma unroll
                    for (int q = 0; q < 4; ++q) {
                        float v = f * (float)(1 << q);
                        F[48 + a*4 + q] = (_Float16)__sinf(v);
                        F[60 + a*4 + q] = (_Float16)__cosf(v);
                    }
                }
                for (int k = 72; k < 96; ++k) F[k] = (_Float16)0.0f;  // K padding
            }
        }
        __syncthreads();

        // Layer 0: [64x96] @ [96x128] -> relu -> sB (f16)
        {
            int m0 = (wave & 3) * 16;
            v16h a0 = load_afrag_lds(sA, m0,  0, lane);
            v16h a1 = load_afrag_lds(sA, m0, 32, lane);
            v16h a2 = load_afrag_lds(sA, m0, 64, lane);
            int nt0 = (wave >> 2) * 4;
            #pragma unroll
            for (int nt = 0; nt < 4; ++nt) {
                int nbase = (nt0 + nt) * 16;
                v8f acc = {};
                acc = wmma_f16(a0, load_bfrag(sW0, nbase,  0, 96, lane), acc);
                acc = wmma_f16(a1, load_bfrag(sW0, nbase, 32, 96, lane), acc);
                acc = wmma_f16(a2, load_bfrag(sW0, nbase, 64, 96, lane), acc);
                int col = nbase + (lane & 15);
                float bv = sBias0[col];
                int rbase = m0 + ((lane >> 4) << 3);
                #pragma unroll
                for (int r = 0; r < 8; ++r)
                    sB[(rbase + r) * 128 + col] = (_Float16)fmaxf(acc[r] + bv, 0.0f);
            }
        }
        __syncthreads();

        // Layer 1: [64x128] @ [128x128] -> relu -> sA (f16)
        {
            int m0 = (wave & 3) * 16;
            v16h a0 = load_afrag_lds(sB, m0,  0, lane);
            v16h a1 = load_afrag_lds(sB, m0, 32, lane);
            v16h a2 = load_afrag_lds(sB, m0, 64, lane);
            v16h a3 = load_afrag_lds(sB, m0, 96, lane);
            int nt0 = (wave >> 2) * 4;
            #pragma unroll
            for (int nt = 0; nt < 4; ++nt) {
                int nbase = (nt0 + nt) * 16;
                v8f acc = {};
                acc = wmma_f16(a0, load_bfrag(sW1, nbase,  0, 128, lane), acc);
                acc = wmma_f16(a1, load_bfrag(sW1, nbase, 32, 128, lane), acc);
                acc = wmma_f16(a2, load_bfrag(sW1, nbase, 64, 128, lane), acc);
                acc = wmma_f16(a3, load_bfrag(sW1, nbase, 96, 128, lane), acc);
                int col = nbase + (lane & 15);
                float bv = sBias1[col];
                int rbase = m0 + ((lane >> 4) << 3);
                #pragma unroll
                for (int r = 0; r < 8; ++r)
                    sA[(rbase + r) * 128 + col] = (_Float16)fmaxf(acc[r] + bv, 0.0f);
            }
        }
        __syncthreads();

        // Layer 2: [64x128] @ [128x16] -> sigmoid/mask/weight -> accumulate
        if (wave < 4) {
            int m0 = wave * 16;
            v8f acc = {};
            #pragma unroll
            for (int kk = 0; kk < 4; ++kk) {
                v16h a = load_afrag_lds(sA, m0, kk * 32, lane);
                v16h b = load_bfrag(sW2, 0, kk * 32, 128, lane);
                acc = wmma_f16(a, b, acc);
            }
            int col = lane & 15;
            if (col < 3) {
                float bv = b2[col];
                int rbase = m0 + ((lane >> 4) << 3);
                #pragma unroll
                for (int r = 0; r < 8; ++r) {
                    int s = ch * 64 + rbase + r;
                    float w = sWt[s];
                    float logit = acc[r] + bv;
                    /* reference: logit zeroed where !mask, then sigmoid -> 0.5 */
                    float rgb = (w > THRES_F) ? (1.0f / (1.0f + __expf(-logit))) : 0.5f;
                    atomicAdd(&sAcc[col], w * rgb);
                }
            }
        }
        __syncthreads();
    }

    if (tid < 3)
        out[ray * 3 + tid] = sAcc[tid] + sLast * BG_F;
}

// ---------------------------------------------------------------------------
extern "C" void kernel_launch(void* const* d_in, const int* in_sizes, int n_in,
                              void* d_out, int out_size, void* d_ws, size_t ws_size,
                              hipStream_t stream)
{
    (void)in_sizes; (void)n_in; (void)out_size; (void)ws_size;
    const float* rays_o  = (const float*)d_in[0];
    const float* rays_d  = (const float*)d_in[1];
    const float* density = (const float*)d_in[2];
    const float* k0      = (const float*)d_in[3];
    const float* w0      = (const float*)d_in[4];
    const float* b0      = (const float*)d_in[5];
    const float* w1      = (const float*)d_in[6];
    const float* b1      = (const float*)d_in[7];
    const float* w2      = (const float*)d_in[8];
    const float* b2      = (const float*)d_in[9];
    _Float16* ws = (_Float16*)d_ws;

    pack_weights_kernel<<<(PACK_TOTAL + 255) / 256, 256, 0, stream>>>(w0, w1, w2, ws);

    render_kernel<<<NRAYS, 256, 0, stream>>>(rays_o, rays_d, density, k0,
                                             ws + W0T_OFF, b0,
                                             ws + W1T_OFF, b1,
                                             ws + W2T_OFF, b2,
                                             (float*)d_out);
}
